// VectorQuantizer_3435973837536
// MI455X (gfx1250) — matrix operand order repaired and checked by reading, not compiled, not run on hardware
//
#include <hip/hip_runtime.h>
#include <hip/hip_bf16.h>

// VQ forward on MI455X (gfx1250):
//   d[n,k] = ||z_n||^2 - 2 z_n.e_k + ||e_k||^2 ; argmin_k ; q = emb[idx] ;
//   out = q (BCHW) ; loss = 1.25 * mean((q-z)^2)
// Distance matmul: v_wmma_f32_16x16x32_bf16 with split-precision compensation
// (z_hi*e_hi + z_lo*e_hi + z_hi*e_lo). Codebook chunks are streamed into
// double-buffered LDS with the Tensor Data Mover (tensor_load_to_lds,
// TENSORcnt), overlapping DMA with WMMA compute.

typedef __attribute__((ext_vector_type(16))) __bf16 v16bf;
typedef __attribute__((ext_vector_type(8)))  float  v8f;
typedef unsigned int u32x4 __attribute__((ext_vector_type(4)));
typedef int          i32x8 __attribute__((ext_vector_type(8)));
typedef int          i32x4 __attribute__((ext_vector_type(4)));

#define VQ_B   16
#define VQ_C   64
#define VQ_HW  4096          // 64*64
#define VQ_N   65536         // B*HW
#define VQ_K   1024
#define CHUNK  128           // codes per LDS-staged chunk
#define NCHUNK (VQ_K / CHUNK)
#define CHUNK_BF16 (2 * CHUNK * VQ_C)   // hi + lo halves: 16384 bf16 = 32 KB

// ---------------------------------------------------------------- prep ------
// Split codebook into bf16 hi/lo in TDM-friendly chunked layout:
//   cb[chunk] = [ hi: 128x64 bf16 (16KB) | lo: 128x64 bf16 (16KB) ]
__global__ void vq_prep(const float* __restrict__ emb,
                        __bf16* __restrict__ cb, float* __restrict__ e2) {
  int k = blockIdx.x * blockDim.x + threadIdx.x;
  if (k >= VQ_K) return;
  int chunk = k >> 7, kl = k & (CHUNK - 1);
  __bf16* hi = cb + chunk * CHUNK_BF16 + kl * VQ_C;
  __bf16* lo = hi + CHUNK * VQ_C;
  float s = 0.f;
  #pragma unroll 4
  for (int c = 0; c < VQ_C; ++c) {
    float v = emb[k * VQ_C + c];
    __bf16 h = (__bf16)v;
    hi[c] = h;
    lo[c] = (__bf16)(v - (float)h);
    s += v * v;
  }
  e2[k] = s;
}

// -------------------------------------------------------------- TDM ---------
// One descriptor moves a whole 32KB chunk (4096 x 8B, 1-D tile) to LDS.
// D# layout per CDNA5 ISA (groups little-endian packed).
// amdgpu-toolchain (clang-23) 6-arg builtin form.
__device__ __forceinline__ void tdm_load_chunk(const __bf16* gsrc, unsigned ldsAddr) {
  unsigned long long ga = (unsigned long long)(uintptr_t)gsrc;
  u32x4 g0;
  g0[0] = 1u;                                          // count=1 (valid), user
  g0[1] = ldsAddr;                                     // lds_addr [63:32]
  g0[2] = (unsigned)(ga & 0xffffffffu);                // global_addr lo
  g0[3] = (unsigned)((ga >> 32) & 0x01ffffffu)         // global_addr [56:32]
        | (2u << 30);                                  // type=2 ("image")
  i32x8 g1;
  g1[0] = (int)(3u << 16);        // wg_mask=0, data_size=3 (8B), no extras
  g1[1] = (int)(4096u << 16);     // tensor_dim0[15:0]=4096 @ bits[63:48]
  g1[2] = (int)(1u << 16);        // tensor_dim0 hi=0 ; tensor_dim1=1
  g1[3] = (int)(4096u << 16);     // tensor_dim1 hi=0 ; tile_dim0=4096
  g1[4] = 1;                      // tile_dim1=1 ; tile_dim2=0
  g1[5] = 4096;                   // tensor_dim0_stride lo32
  g1[6] = 0;                      // stride hi ; tensor_dim1_stride lo
  g1[7] = 0;
  i32x4 gz4 = {0, 0, 0, 0};                 // <=2D: groups 2/3 unused
  i32x8 gz8 = {0, 0, 0, 0, 0, 0, 0, 0};     // extra group (6-arg form)
  __builtin_amdgcn_tensor_load_to_lds(g0, g1, gz4, gz4, gz8, 0);
}

// ------------------------------------------------------------- distance -----
// A fragment (16x32 bf16, MxK): lane<16 -> row=lane, elems [0..15]=K0..15 ;
// lane>=16 -> row=lane-16, elems [0..15]=K16..31.
__device__ __forceinline__ void load_a_frag(const float* __restrict__ z,
                                            int rowBase, int lane,
                                            v16bf& hi0, v16bf& hi1,
                                            v16bf& lo0, v16bf& lo1) {
  int r  = rowBase + (lane & 15);
  int b  = r >> 12;            // HW = 4096
  int hw = r & 4095;
  int off = (lane < 16) ? 0 : 16;
  const float* base = z + (size_t)b * VQ_C * VQ_HW + hw;
  #pragma unroll
  for (int e = 0; e < 16; ++e) {
    int cb = off + e;                           // K within a 32-wide chunk
    float f0 = base[(size_t)(cb)      * VQ_HW]; // ka = 0 (c = cb)
    float f1 = base[(size_t)(32 + cb) * VQ_HW]; // ka = 1 (c = 32+cb)
    __bf16 h0 = (__bf16)f0, h1 = (__bf16)f1;
    hi0[e] = h0;  lo0[e] = (__bf16)(f0 - (float)h0);
    hi1[e] = h1;  lo1[e] = (__bf16)(f1 - (float)h1);
  }
}

// B fragment (32x16 bf16, KxN): lane<16 -> col=lane, K=0..15 ;
// lane>=16 -> col=lane-16, K=16..31. 16 consecutive bf16 (32B) per lane.
__device__ __forceinline__ v16bf load_b_frag(const __bf16* sh,
                                             int codeLocal, int lane, int ka) {
  int cb = ka * 32 + ((lane < 16) ? 0 : 16);
  return *(const v16bf*)(sh + codeLocal * VQ_C + cb);
}

__global__ void __launch_bounds__(256)
vq_dist(const float* __restrict__ z, const __bf16* __restrict__ cb,
        const float* __restrict__ e2g, int* __restrict__ idx) {
  // Double-buffered chunk staging: 2 x 32KB (hi|lo) = 64KB of 320KB/WGP.
  __shared__ __align__(32) __bf16 sh[2][CHUNK_BF16];

  const int lane    = threadIdx.x & 31;
  const int wave    = threadIdx.x >> 5;               // 8 waves / block
  const int rowBase = (blockIdx.x * 8 + wave) * 16;   // 16-row tile / wave

  // z rows -> registers once; reused across all 1024 codes.
  v16bf ahi0, ahi1, alo0, alo1;
  load_a_frag(z, rowBase, lane, ahi0, ahi1, alo0, alo1);

  float minv[8];
  int   mini[8];
  #pragma unroll
  for (int r = 0; r < 8; ++r) { minv[r] = 3.4e38f; mini[r] = 0; }

  const unsigned lds0 = (unsigned)(uintptr_t)&sh[0][0];
  const unsigned lds1 = (unsigned)(uintptr_t)&sh[1][0];

  if (threadIdx.x == 0) {                 // wave 0 drives the TDM pipeline
    tdm_load_chunk(cb, lds0);
    __builtin_amdgcn_s_wait_tensorcnt(0);
  }
  __syncthreads();

  for (int chunk = 0; chunk < NCHUNK; ++chunk) {
    // Kick DMA of the next chunk into the other buffer while computing.
    if (threadIdx.x == 0 && chunk + 1 < NCHUNK)
      tdm_load_chunk(cb + (chunk + 1) * CHUNK_BF16,
                     ((chunk + 1) & 1) ? lds1 : lds0);

    const __bf16* shh = sh[chunk & 1];                  // hi half
    const __bf16* shl = sh[chunk & 1] + CHUNK * VQ_C;   // lo half

    for (int tl = 0; tl < CHUNK / 16; ++tl) {
      const int cl = tl * 16 + (lane & 15);             // local code (col)
      v16bf bh0 = load_b_frag(shh, cl, lane, 0);
      v16bf bh1 = load_b_frag(shh, cl, lane, 1);
      v16bf bl0 = load_b_frag(shl, cl, lane, 0);
      v16bf bl1 = load_b_frag(shl, cl, lane, 1);

      v8f acc = {};
      acc = __builtin_amdgcn_wmma_f32_16x16x32_bf16(false, ahi0, false, bh0, (short)0, acc, false, false);
      acc = __builtin_amdgcn_wmma_f32_16x16x32_bf16(false, ahi1, false, bh1, (short)0, acc, false, false);
      acc = __builtin_amdgcn_wmma_f32_16x16x32_bf16(false, alo0, false, bh0, (short)0, acc, false, false);
      acc = __builtin_amdgcn_wmma_f32_16x16x32_bf16(false, alo1, false, bh1, (short)0, acc, false, false);
      acc = __builtin_amdgcn_wmma_f32_16x16x32_bf16(false, ahi0, false, bl0, (short)0, acc, false, false);
      acc = __builtin_amdgcn_wmma_f32_16x16x32_bf16(false, ahi1, false, bl1, (short)0, acc, false, false);

      const int   code = chunk * CHUNK + cl;
      const float ev   = e2g[code];                     // L1/L2-resident
      #pragma unroll
      for (int r = 0; r < 8; ++r) {
        float d = ev - 2.0f * acc[r];   // + ||z||^2 omitted (row-constant)
        if (d < minv[r]) { minv[r] = d; mini[r] = code; }
      }
    }

    __syncthreads();                    // all waves done reading sh[chunk&1]
    if (threadIdx.x == 0)
      __builtin_amdgcn_s_wait_tensorcnt(0);   // next buffer landed
    __syncthreads();                    // publish LDS to all waves
  }

  // Reduce argmin across the 16 lanes holding different codes (keep halves
  // separate: lanes 0-15 own rows 0-7, lanes 16-31 own rows 8-15).
  #pragma unroll
  for (int r = 0; r < 8; ++r) {
    float v = minv[r];
    int   i = mini[r];
    #pragma unroll
    for (int m = 1; m <= 8; m <<= 1) {
      float ov = __shfl_xor(v, m, 32);
      int   oi = __shfl_xor(i, m, 32);
      if (ov < v || (ov == v && oi < i)) { v = ov; i = oi; }  // first-match tie
    }
    minv[r] = v; mini[r] = i;
  }
  if ((lane & 15) == 0) {
    int rb = rowBase + ((lane >> 4) ? 8 : 0);
    #pragma unroll
    for (int r = 0; r < 8; ++r) idx[rb + r] = mini[r];
  }
}

// ----------------------------------------------------- gather + loss --------
__global__ void __launch_bounds__(256)
vq_gather_loss(const float* __restrict__ z, const float* __restrict__ emb,
               const int* __restrict__ idx, float* __restrict__ out,
               float* __restrict__ partials) {
  __shared__ float red[256];
  int tid = blockIdx.x * 256 + threadIdx.x;   // linear BCHW index
  int b  = tid >> 18;                         // C*HW = 262144
  int c  = (tid >> 12) & 63;
  int hw = tid & 4095;
  int n  = (b << 12) | hw;
  float q  = emb[idx[n] * VQ_C + c];
  float zz = z[tid];
  out[tid] = q;                               // forward(q_st) == q
  float d = q - zz;
  red[threadIdx.x] = d * d;
  __syncthreads();
  #pragma unroll
  for (int s = 128; s > 0; s >>= 1) {
    if (threadIdx.x < s) red[threadIdx.x] += red[threadIdx.x + s];
    __syncthreads();
  }
  if (threadIdx.x == 0) partials[blockIdx.x] = red[0];
}

__global__ void vq_final(const float* __restrict__ partials, float* __restrict__ loss) {
  __shared__ float red[256];
  float s = 0.f;
  for (int i = threadIdx.x; i < 16384; i += 256) s += partials[i];
  red[threadIdx.x] = s;
  __syncthreads();
  #pragma unroll
  for (int st = 128; st > 0; st >>= 1) {
    if (threadIdx.x < st) red[threadIdx.x] += red[threadIdx.x + st];
    __syncthreads();
  }
  // loss = q_latent + 0.25*e_latent = 1.25 * mean((q-z)^2)
  if (threadIdx.x == 0) loss[0] = 1.25f * red[0] / 4194304.0f;
}

// ---------------------------------------------------------------- launch ----
extern "C" void kernel_launch(void* const* d_in, const int* in_sizes, int n_in,
                              void* d_out, int out_size, void* d_ws, size_t ws_size,
                              hipStream_t stream) {
  const float* z   = (const float*)d_in[0];   // [16,64,64,64]
  const float* emb = (const float*)d_in[1];   // [1024,64]
  float* out = (float*)d_out;                 // 4194304 q values + 1 loss

  char* ws = (char*)d_ws;
  __bf16* wcb  = (__bf16*)(ws);               // 256 KB chunked codebook hi|lo
  float*  we2  = (float*) (ws + 262144);      // 4 KB ||e||^2
  int*    widx = (int*)   (ws + 266240);      // 256 KB argmin indices
  float*  wpar = (float*) (ws + 528384);      // 64 KB block partial sums

  vq_prep       <<<4,     256, 0, stream>>>(emb, wcb, we2);
  vq_dist       <<<512,   256, 0, stream>>>(z, wcb, we2, widx);
  vq_gather_loss<<<16384, 256, 0, stream>>>(z, emb, widx, out, wpar);
  vq_final      <<<1,     256, 0, stream>>>(wpar, out + 4194304);
}
